// HeteroModel_77421080477844
// MI455X (gfx1250) — matrix operand order repaired and checked by reading, not compiled, not run on hardware
//
#include <hip/hip_runtime.h>
#include <stddef.h>

// ---------------------------------------------------------------------------
// HeteroGNN (GraphConv x2 + link predictor) for MI455X / gfx1250.
//  - GEMMs via v_wmma_f32_16x16x32_bf16 (bf16 A/B staged in LDS / L2, f32 acc)
//  - segment_sum via native global_atomic_add_f32 (dest buffers L2-resident;
//    forced via inline asm so LLVM cannot expand to a CAS loop)
//  - final dot via coalesced float4 gathers + wave32 shfl reduction
// ---------------------------------------------------------------------------

typedef __bf16  v16bf __attribute__((ext_vector_type(16)));
typedef __bf16  v8bf  __attribute__((ext_vector_type(8)));
typedef float   v8f   __attribute__((ext_vector_type(8)));

// ---------------- weight convert + transpose: W[K][N] f32 -> Wt[N][K] bf16 --
__global__ __launch_bounds__(256) void convert_transpose(
    const float* __restrict__ W, __bf16* __restrict__ Wt, int K, int N) {
  int idx = blockIdx.x * 256 + threadIdx.x;
  if (idx < K * N) {
    int k = idx / N, n = idx % N;
    Wt[(size_t)n * K + k] = (__bf16)W[idx];
  }
}

// ---------------- zero a buffer in float4 units ----------------------------
__global__ __launch_bounds__(256) void zero_f32(float* __restrict__ p, int n4) {
  int idx = blockIdx.x * 256 + threadIdx.x;
  if (idx < n4) ((float4*)p)[idx] = make_float4(0.f, 0.f, 0.f, 0.f);
}

// ---------------- edge scatter-add: dst[didx[e]] += src[gidx[e]] (H=128) ---
// Native CDNA5 no-return FP32 atomic (L2 performs the RMW). Inline asm
// guarantees GLOBAL_ATOMIC_ADD_F32 instead of a cmpswap expansion.
__device__ __forceinline__ void atom_add_f32(float* p, float v) {
  asm volatile("global_atomic_add_f32 %0, %1, off"
               :
               : "v"(p), "v"(v)
               : "memory");
}

__global__ __launch_bounds__(256) void scatter_add_rows(
    float* __restrict__ dst, const float* __restrict__ src,
    const int* __restrict__ gidx, const int* __restrict__ didx, int E) {
  int gt = blockIdx.x * 256 + threadIdx.x;
  int e = gt >> 5;
  int lane = gt & 31;
  if (e >= E) return;
  int s = gidx[e];
  int d = didx[e];
  const float4 v = *(const float4*)(src + (size_t)s * 128 + lane * 4);
  float* dp = dst + (size_t)d * 128 + lane * 4;
  atom_add_f32(dp + 0, v.x);
  atom_add_f32(dp + 1, v.y);
  atom_add_f32(dp + 2, v.z);
  atom_add_f32(dp + 3, v.w);
}

// ---------------- WMMA GEMM: Y[M,128] = A1@W1 (+ A2@W2) + bias -------------
// A terms are f32 row-major [M,K]; W terms are bf16 transposed [128][K].
// Block: 256 threads (8 waves). Each block computes a 32x128 output tile;
// each wave owns one 16-column N-tile with two 16x16 accumulators.
__device__ __forceinline__ v16bf make_frag(const __bf16* p0, const __bf16* p1) {
  v8bf lo = *(const v8bf*)p0;
  v8bf hh = *(const v8bf*)p1;
  v16bf r;
#pragma unroll
  for (int i = 0; i < 8; ++i) { r[i] = lo[i]; r[i + 8] = hh[i]; }
  return r;
}

__global__ __launch_bounds__(256) void wmma_gemm_bias(
    const float* __restrict__ A1, int K1, const __bf16* __restrict__ Wt1,
    const float* __restrict__ A2, int K2, const __bf16* __restrict__ Wt2,
    const float* __restrict__ bias, float* __restrict__ Y,
    int M, int nterms) {
  __shared__ __align__(32) __bf16 Asm[2 * 32 * 128];  // 16 KB
  const int tid = threadIdx.x;
  const int m0 = blockIdx.x * 32;

  // stage A tiles (f32 -> bf16), zero-fill beyond K / beyond M
  for (int t = 0; t < nterms; ++t) {
    const float* __restrict__ A = (t == 0) ? A1 : A2;
    const int K = (t == 0) ? K1 : K2;
    for (int idx = tid; idx < 32 * 128; idx += 256) {
      const int r = idx >> 7, c = idx & 127;
      const int m = m0 + r;
      float v = (c < K && m < M) ? A[(size_t)m * K + c] : 0.0f;
      Asm[t * 4096 + idx] = (__bf16)v;
    }
  }
  __syncthreads();

  const int lane = tid & 31;
  const int wid  = tid >> 5;     // N-tile 0..7
  const int hi   = lane >> 4;    // half-wave
  const int mrow = lane & 15;
  const int col  = wid * 16 + mrow;   // 0..127

  const float bv = bias[col];
  v8f acc0, acc1;
#pragma unroll
  for (int i = 0; i < 8; ++i) { acc0[i] = bv; acc1[i] = bv; }

  for (int t = 0; t < nterms; ++t) {
    const int K = (t == 0) ? K1 : K2;
    const __bf16* __restrict__ wcol = ((t == 0) ? Wt1 : Wt2) + (size_t)col * K;
    const __bf16* a0 = Asm + t * 4096 + mrow * 128;
    const __bf16* a1 = Asm + t * 4096 + (16 + mrow) * 128;
    for (int kk = 0; kk < K; kk += 32) {
      // B fragment: lanes 0-15 hold K kk..kk+15, lanes 16-31 hold kk+16..kk+31
      // of column `col` -> contiguous 32B in the transposed weight.
      const v16bf b = *(const v16bf*)(wcol + kk + hi * 16);
      // A fragment: lane<16 -> K kk..kk+15;
      //             lane>=16 -> K kk+16..kk+31
      const v16bf f0 = make_frag(a0 + kk + hi * 16, a0 + kk + hi * 16 + 8);
      const v16bf f1 = make_frag(a1 + kk + hi * 16, a1 + kk + hi * 16 + 8);
      acc0 = __builtin_amdgcn_wmma_f32_16x16x32_bf16(
          false, f0, false, b, (short)0, acc0, false, false);
      acc1 = __builtin_amdgcn_wmma_f32_16x16x32_bf16(
          false, f1, false, b, (short)0, acc1, false, false);
    }
  }

  // C/D layout: VGPR j -> (M = j + 8*hi, N = col)
#pragma unroll
  for (int j = 0; j < 8; ++j) {
    const int r0 = m0 + j + 8 * hi;
    if (r0 < M) Y[(size_t)r0 * 128 + col] = acc0[j];
    const int r1 = m0 + 16 + j + 8 * hi;
    if (r1 < M) Y[(size_t)r1 * 128 + col] = acc1[j];
  }
}

// ---------------- link predictor: out[e] = dot(xt[ls[e]], xp[ld[e]]) -------
__global__ __launch_bounds__(256) void link_dot(
    const float* __restrict__ xt, const float* __restrict__ xp,
    const int* __restrict__ ls, const int* __restrict__ ld,
    float* __restrict__ out, int EL) {
  int gt = blockIdx.x * 256 + threadIdx.x;
  int e = gt >> 5;
  int lane = gt & 31;
  if (e >= EL) return;
  int a = ls[e];
  int b = ld[e];
  const float4 u = *(const float4*)(xt + (size_t)a * 128 + lane * 4);
  const float4 v = *(const float4*)(xp + (size_t)b * 128 + lane * 4);
  float s = u.x * v.x + u.y * v.y + u.z * v.z + u.w * v.w;
#pragma unroll
  for (int m = 16; m >= 1; m >>= 1) s += __shfl_xor(s, m, 32);
  if (lane == 0) out[e] = s;
}

// ---------------------------------------------------------------------------
extern "C" void kernel_launch(void* const* d_in, const int* in_sizes, int n_in,
                              void* d_out, int out_size, void* d_ws, size_t ws_size,
                              hipStream_t stream) {
  (void)in_sizes; (void)n_in; (void)out_size; (void)ws_size;
  constexpr int NT = 200000, NP = 50000, H = 128, E = 1000000, EL = 500000, L = 2;

  const float* x_track    = (const float*)d_in[0];
  const float* x_playlist = (const float*)d_in[1];
  const int*   edge_src   = (const int*)d_in[2];
  const int*   edge_dst   = (const int*)d_in[3];
  const int*   label_src  = (const int*)d_in[4];
  const int*   label_dst  = (const int*)d_in[5];
  const float* W_track    = (const float*)d_in[6];
  const float* b_track    = (const float*)d_in[7];
  const float* W_playlist = (const float*)d_in[8];
  const float* b_playlist = (const float*)d_in[9];
  const float* W_rel_cp   = (const float*)d_in[10];
  const float* b_rel_cp   = (const float*)d_in[11];
  const float* W_root_cp  = (const float*)d_in[12];
  const float* W_rel_rc   = (const float*)d_in[13];
  const float* b_rel_rc   = (const float*)d_in[14];
  const float* W_root_rc  = (const float*)d_in[15];

  // workspace carve-out (256B aligned): ~385 MB
  char* ws = (char*)d_ws;
  size_t off = 0;
  auto take = [&](size_t bytes) -> void* {
    void* p = (void*)(ws + off);
    off += (bytes + 255) & ~(size_t)255;
    return p;
  };
  float* xt[2] = {(float*)take((size_t)NT * H * 4), (float*)take((size_t)NT * H * 4)};
  float* xp[2] = {(float*)take((size_t)NP * H * 4), (float*)take((size_t)NP * H * 4)};
  float* msg_t = (float*)take((size_t)NT * H * 4);
  float* msg_p = (float*)take((size_t)NP * H * 4);
  __bf16* wt_track = (__bf16*)take(128 * 128 * 2);
  __bf16* wt_play  = (__bf16*)take(128 * 64 * 2);
  __bf16* wt_rel_cp[L], *wt_root_cp[L], *wt_rel_rc[L], *wt_root_rc[L];
  for (int l = 0; l < L; ++l) {
    wt_rel_cp[l]  = (__bf16*)take(128 * 128 * 2);
    wt_root_cp[l] = (__bf16*)take(128 * 128 * 2);
    wt_rel_rc[l]  = (__bf16*)take(128 * 128 * 2);
    wt_root_rc[l] = (__bf16*)take(128 * 128 * 2);
  }

  // 1) convert + transpose all weights to bf16 [N][K]
  auto conv = [&](const float* W, __bf16* Wt, int K, int N) {
    int n = K * N;
    convert_transpose<<<(n + 255) / 256, 256, 0, stream>>>(W, Wt, K, N);
  };
  conv(W_track, wt_track, 128, 128);
  conv(W_playlist, wt_play, 64, 128);
  for (int l = 0; l < L; ++l) {
    conv(W_rel_cp  + (size_t)l * 128 * 128, wt_rel_cp[l],  128, 128);
    conv(W_root_cp + (size_t)l * 128 * 128, wt_root_cp[l], 128, 128);
    conv(W_rel_rc  + (size_t)l * 128 * 128, wt_rel_rc[l],  128, 128);
    conv(W_root_rc + (size_t)l * 128 * 128, wt_root_rc[l], 128, 128);
  }

  // 2) input projections
  wmma_gemm_bias<<<(NT + 31) / 32, 256, 0, stream>>>(
      x_track, 128, wt_track, nullptr, 0, nullptr, b_track, xt[0], NT, 1);
  wmma_gemm_bias<<<(NP + 31) / 32, 256, 0, stream>>>(
      x_playlist, 64, wt_play, nullptr, 0, nullptr, b_playlist, xp[0], NP, 1);

  // 3) GraphConv layers
  int cur = 0;
  for (int l = 0; l < L; ++l) {
    zero_f32<<<(NP * 32 + 255) / 256, 256, 0, stream>>>(msg_p, NP * 32);
    zero_f32<<<(NT * 32 + 255) / 256, 256, 0, stream>>>(msg_t, NT * 32);
    // track -> playlist
    scatter_add_rows<<<(E * 32 + 255) / 256, 256, 0, stream>>>(
        msg_p, xt[cur], edge_src, edge_dst, E);
    // playlist -> track
    scatter_add_rows<<<(E * 32 + 255) / 256, 256, 0, stream>>>(
        msg_t, xp[cur], edge_dst, edge_src, E);
    // xp_new = msg_p @ W_rel_cp + b + xp @ W_root_cp
    wmma_gemm_bias<<<(NP + 31) / 32, 256, 0, stream>>>(
        msg_p, 128, wt_rel_cp[l], xp[cur], 128, wt_root_cp[l],
        b_rel_cp + (size_t)l * H, xp[1 - cur], NP, 2);
    // xt_new = msg_t @ W_rel_rc + b + xt @ W_root_rc
    wmma_gemm_bias<<<(NT + 31) / 32, 256, 0, stream>>>(
        msg_t, 128, wt_rel_rc[l], xt[cur], 128, wt_root_rc[l],
        b_rel_rc + (size_t)l * H, xt[1 - cur], NT, 2);
    cur = 1 - cur;
  }

  // 4) link predictor
  link_dot<<<(EL * 32 + 255) / 256, 256, 0, stream>>>(
      xt[cur], xp[cur], label_src, label_dst, (float*)d_out, EL);
}
